// GAT_44263932952821
// MI455X (gfx1250) — matrix operand order repaired and checked by reading, not compiled, not run on hardware
//
#include <hip/hip_runtime.h>

typedef __attribute__((ext_vector_type(16))) _Float16 v16h;
typedef __attribute__((ext_vector_type(8)))  float    v8f;

#define NEG_SLOPE 0.2f

// ---------------- utility ----------------
__global__ void k_zero(float* __restrict__ p, long n) {
  long i = (long)blockIdx.x * blockDim.x + threadIdx.x;
  if (i < n) p[i] = 0.0f;
}

// ---------------- GEMM1: h1 = x @ W1, [N,128]x[128,64], WMMA f16->f32 ----
// W1 is pre-swizzled in LDS into per-lane WMMA B-fragment order so each lane
// reads its whole 16-half fragment as one contiguous 32-byte LDS load.
__global__ void __launch_bounds__(256) k_gemm1(const float* __restrict__ x,
                                               const float* __restrict__ W1,
                                               float* __restrict__ h1, int N) {
  __shared__ v16h w1p[16 * 32];            // 16 fragments (kt 0..3, nt 0..3) x 32 lanes
  _Float16* wph = (_Float16*)w1p;
  int tid = threadIdx.x;
  for (int i = tid; i < 16 * 32 * 16; i += 256) {
    int idx  = i & 15;                     // element within lane fragment
    int lane = (i >> 4) & 31;
    int frag = i >> 9;                     // kt*4 + nt
    int kt = frag >> 2, nt = frag & 3;
    int hi2 = lane >> 4, ln2 = lane & 15;
    int j = idx >> 1, lo = idx & 1;
    int kr = kt * 32 + hi2 * 8 + (idx >> 3) * 16 + (idx & 7);
    wph[i] = (_Float16)W1[kr * 64 + nt * 16 + ln2];
  }
  __syncthreads();

  int wave = tid >> 5, lane = tid & 31;
  int hi = (lane >> 4) & 1, ln = lane & 15;
  int rowBase = blockIdx.x * 128 + wave * 16;
  int m = rowBase + ln;
  const float* xrow = x + (size_t)(m < N ? m : 0) * 128;

  v8f acc[4] = {};
#pragma unroll
  for (int kt = 0; kt < 4; ++kt) {
    int K0 = kt * 32;
    // two contiguous 8-float runs per lane -> 4x b128 loads + packed converts
    const float4* p0 = (const float4*)(xrow + K0 + hi * 8);
    const float4* p1 = (const float4*)(xrow + K0 + 16 + hi * 8);
    float4 r0 = p0[0], r1 = p0[1], r2 = p1[0], r3 = p1[1];
    v16h a;
    a[0]  = (_Float16)r0.x; a[1]  = (_Float16)r0.y;
    a[2]  = (_Float16)r0.z; a[3]  = (_Float16)r0.w;
    a[4]  = (_Float16)r1.x; a[5]  = (_Float16)r1.y;
    a[6]  = (_Float16)r1.z; a[7]  = (_Float16)r1.w;
    a[8]  = (_Float16)r2.x; a[9]  = (_Float16)r2.y;
    a[10] = (_Float16)r2.z; a[11] = (_Float16)r2.w;
    a[12] = (_Float16)r3.x; a[13] = (_Float16)r3.y;
    a[14] = (_Float16)r3.z; a[15] = (_Float16)r3.w;
#pragma unroll
    for (int nt = 0; nt < 4; ++nt) {
      v16h b = w1p[(kt * 4 + nt) * 32 + lane];
      acc[nt] = __builtin_amdgcn_wmma_f32_16x16x32_f16(
          false, a, false, b, (short)0, acc[nt], false, false);
    }
  }
#pragma unroll
  for (int nt = 0; nt < 4; ++nt)
#pragma unroll
    for (int r = 0; r < 8; ++r) {
      int row = rowBase + r + hi * 8;
      if (row < N) h1[(size_t)row * 64 + nt * 16 + ln] = acc[nt][r];
    }
}

// ---------------- GEMM2: h2 = h @ W2, [N,64]x[64,C<=48], WMMA ------------
__global__ void __launch_bounds__(256) k_gemm2(const float* __restrict__ h,
                                               const float* __restrict__ W2,
                                               float* __restrict__ h2, int N, int C) {
  __shared__ v16h w2p[6 * 32];             // 6 fragments (kt 0..1, nt 0..2) x 32 lanes
  _Float16* wph = (_Float16*)w2p;
  int tid = threadIdx.x;
  for (int i = tid; i < 6 * 32 * 16; i += 256) {
    int idx  = i & 15;
    int lane = (i >> 4) & 31;
    int frag = i >> 9;                     // kt*3 + nt
    int kt = frag / 3, nt = frag % 3;
    int hi2 = lane >> 4, ln2 = lane & 15;
    int j = idx >> 1, lo = idx & 1;
    int kr  = kt * 32 + hi2 * 8 + (idx >> 3) * 16 + (idx & 7);
    int col = nt * 16 + ln2;
    wph[i] = (col < C) ? (_Float16)W2[kr * C + col] : (_Float16)0.0f;
  }
  __syncthreads();

  int wave = tid >> 5, lane = tid & 31;
  int hi = (lane >> 4) & 1, ln = lane & 15;
  int rowBase = blockIdx.x * 128 + wave * 16;
  int m = rowBase + ln;
  const float* hrow = h + (size_t)(m < N ? m : 0) * 64;

  v8f acc[3] = {};
#pragma unroll
  for (int kt = 0; kt < 2; ++kt) {
    int K0 = kt * 32;
    const float4* p0 = (const float4*)(hrow + K0 + hi * 8);
    const float4* p1 = (const float4*)(hrow + K0 + 16 + hi * 8);
    float4 r0 = p0[0], r1 = p0[1], r2 = p1[0], r3 = p1[1];
    v16h a;
    a[0]  = (_Float16)r0.x; a[1]  = (_Float16)r0.y;
    a[2]  = (_Float16)r0.z; a[3]  = (_Float16)r0.w;
    a[4]  = (_Float16)r1.x; a[5]  = (_Float16)r1.y;
    a[6]  = (_Float16)r1.z; a[7]  = (_Float16)r1.w;
    a[8]  = (_Float16)r2.x; a[9]  = (_Float16)r2.y;
    a[10] = (_Float16)r2.z; a[11] = (_Float16)r2.w;
    a[12] = (_Float16)r3.x; a[13] = (_Float16)r3.y;
    a[14] = (_Float16)r3.z; a[15] = (_Float16)r3.w;
#pragma unroll
    for (int nt = 0; nt < 3; ++nt) {
      v16h b = w2p[(kt * 3 + nt) * 32 + lane];
      acc[nt] = __builtin_amdgcn_wmma_f32_16x16x32_f16(
          false, a, false, b, (short)0, acc[nt], false, false);
    }
  }
#pragma unroll
  for (int nt = 0; nt < 3; ++nt)
#pragma unroll
    for (int r = 0; r < 8; ++r) {
      int row = rowBase + r + hi * 8;
      int col = nt * 16 + ln;
      if (row < N && col < C) h2[(size_t)row * C + col] = acc[nt][r];
    }
}

// ---------------- attention dot products ----------------
__global__ void k_att1(const float* __restrict__ h1, const float* __restrict__ asw,
                       const float* __restrict__ adw, float* __restrict__ as,
                       float* __restrict__ ad, int N) {
  int t = blockIdx.x * blockDim.x + threadIdx.x;
  if (t >= N * 8) return;
  int n = t >> 3, hh = t & 7;
  const float4* hp = (const float4*)(h1 + (size_t)n * 64 + hh * 8);
  float4 v0 = hp[0], v1 = hp[1];
  float vv[8] = {v0.x, v0.y, v0.z, v0.w, v1.x, v1.y, v1.z, v1.w};
  float s = 0.f, d = 0.f;
#pragma unroll
  for (int f = 0; f < 8; ++f) {
    s += vv[f] * asw[hh * 8 + f];
    d += vv[f] * adw[hh * 8 + f];
  }
  as[t] = s;
  ad[t] = d;
}

__global__ void k_att2(const float* __restrict__ h2, const float* __restrict__ asw,
                       const float* __restrict__ adw, float* __restrict__ as,
                       float* __restrict__ ad, int N, int C) {
  int n = blockIdx.x * blockDim.x + threadIdx.x;
  if (n >= N) return;
  const float* hp = h2 + (size_t)n * C;
  float s = 0.f, d = 0.f;
  for (int c = 0; c < C; ++c) {
    float v = hp[c];
    s += v * asw[c];
    d += v * adw[c];
  }
  as[n] = s;
  ad[n] = d;
}

// ---------------- layer-1 edge passes (H=8) ----------------
__global__ void k_esum1(const long long* __restrict__ e0, const long long* __restrict__ e1,
                        int E, int N, const float* __restrict__ as,
                        const float* __restrict__ ad, float* __restrict__ s1) {
  long t = (long)blockIdx.x * blockDim.x + threadIdx.x;
  long tot = (long)(E + N) * 8;
  if (t >= tot) return;
  int e = (int)(t >> 3), hh = (int)(t & 7);
  int src = (e < E) ? (int)e0[e] : e - E;
  int dst = (e < E) ? (int)e1[e] : e - E;
  float v = as[src * 8 + hh] + ad[dst * 8 + hh];
  v = v > 0.f ? v : NEG_SLOPE * v;
  atomicAdd(&s1[dst * 8 + hh], expf(v));
}

__global__ void k_eagg1(const long long* __restrict__ e0, const long long* __restrict__ e1,
                        int E, int N, const float* __restrict__ as,
                        const float* __restrict__ ad, const float* __restrict__ s1,
                        const float* __restrict__ h1, float* __restrict__ agg1) {
  long t = (long)blockIdx.x * blockDim.x + threadIdx.x;
  long tot = (long)(E + N) * 8;
  if (t >= tot) return;
  int e = (int)(t >> 3), hh = (int)(t & 7);
  int src = (e < E) ? (int)e0[e] : e - E;
  int dst = (e < E) ? (int)e1[e] : e - E;
  float v = as[src * 8 + hh] + ad[dst * 8 + hh];
  v = v > 0.f ? v : NEG_SLOPE * v;
  float alpha = expf(v) / (s1[dst * 8 + hh] + 1e-16f);
  const float4* hs = (const float4*)(h1 + (size_t)src * 64 + hh * 8);  // 32B aligned
  float4 v0 = hs[0], v1 = hs[1];
  float* ag = agg1 + (size_t)dst * 64 + hh * 8;
  atomicAdd(&ag[0], v0.x * alpha);
  atomicAdd(&ag[1], v0.y * alpha);
  atomicAdd(&ag[2], v0.z * alpha);
  atomicAdd(&ag[3], v0.w * alpha);
  atomicAdd(&ag[4], v1.x * alpha);
  atomicAdd(&ag[5], v1.y * alpha);
  atomicAdd(&ag[6], v1.z * alpha);
  atomicAdd(&ag[7], v1.w * alpha);
}

// ---------------- elu + bias (writes h into h1's buffer) ----------------
__global__ void k_elu(const float* __restrict__ agg1, const float* __restrict__ b1,
                      float* __restrict__ hout, long n64) {
  long t = (long)blockIdx.x * blockDim.x + threadIdx.x;
  if (t >= n64) return;
  float v = agg1[t] + b1[t & 63];
  hout[t] = v > 0.f ? v : (expf(v) - 1.0f);
}

// ---------------- layer-2 edge passes (H=1, C channels) ----------------
__global__ void k_esum2(const long long* __restrict__ e0, const long long* __restrict__ e1,
                        int E, int N, const float* __restrict__ as,
                        const float* __restrict__ ad, float* __restrict__ s2) {
  long t = (long)blockIdx.x * blockDim.x + threadIdx.x;
  if (t >= (long)(E + N)) return;
  int e = (int)t;
  int src = (e < E) ? (int)e0[e] : e - E;
  int dst = (e < E) ? (int)e1[e] : e - E;
  float v = as[src] + ad[dst];
  v = v > 0.f ? v : NEG_SLOPE * v;
  atomicAdd(&s2[dst], expf(v));
}

__global__ void k_eagg2(const long long* __restrict__ e0, const long long* __restrict__ e1,
                        int E, int N, const float* __restrict__ as,
                        const float* __restrict__ ad, const float* __restrict__ s2,
                        const float* __restrict__ h2, float* __restrict__ agg2,
                        int C, int chunks) {
  long t = (long)blockIdx.x * blockDim.x + threadIdx.x;
  long tot = (long)(E + N) * chunks;
  if (t >= tot) return;
  int e = (int)(t / chunks), ch = (int)(t % chunks);
  int src = (e < E) ? (int)e0[e] : e - E;
  int dst = (e < E) ? (int)e1[e] : e - E;
  float v = as[src] + ad[dst];
  v = v > 0.f ? v : NEG_SLOPE * v;
  float alpha = expf(v) / (s2[dst] + 1e-16f);
  int c0 = ch * 8;
  int c1 = c0 + 8 < C ? c0 + 8 : C;
  const float* hs = h2 + (size_t)src * C;
  float* ag = agg2 + (size_t)dst * C;
  if (c1 - c0 == 8 && (C & 3) == 0) {
    const float4* h4 = (const float4*)(hs + c0);
    float4 v0 = h4[0], v1 = h4[1];
    atomicAdd(&ag[c0 + 0], v0.x * alpha);
    atomicAdd(&ag[c0 + 1], v0.y * alpha);
    atomicAdd(&ag[c0 + 2], v0.z * alpha);
    atomicAdd(&ag[c0 + 3], v0.w * alpha);
    atomicAdd(&ag[c0 + 4], v1.x * alpha);
    atomicAdd(&ag[c0 + 5], v1.y * alpha);
    atomicAdd(&ag[c0 + 6], v1.z * alpha);
    atomicAdd(&ag[c0 + 7], v1.w * alpha);
  } else {
    for (int c = c0; c < c1; ++c) atomicAdd(&ag[c], hs[c] * alpha);
  }
}

// ---------------- output: log_softmax(agg2 + b2) ----------------
__global__ void k_lsm(const float* __restrict__ agg2, const float* __restrict__ b2,
                      float* __restrict__ out, int N, int C) {
  int n = blockIdx.x * blockDim.x + threadIdx.x;
  if (n >= N) return;
  const float* ap = agg2 + (size_t)n * C;
  float v[64];
  float mx = -1e30f;
  for (int c = 0; c < C; ++c) {
    v[c] = ap[c] + b2[c];
    mx = v[c] > mx ? v[c] : mx;
  }
  float s = 0.f;
  for (int c = 0; c < C; ++c) s += expf(v[c] - mx);
  float l = mx + logf(s);
  float* op = out + (size_t)n * C;
  for (int c = 0; c < C; ++c) op[c] = v[c] - l;
}

// ---------------- host launch ----------------
extern "C" void kernel_launch(void* const* d_in, const int* in_sizes, int n_in,
                              void* d_out, int out_size, void* d_ws, size_t ws_size,
                              hipStream_t stream) {
  const float* x        = (const float*)d_in[0];
  const long long* ei   = (const long long*)d_in[1];
  const float* W1       = (const float*)d_in[2];
  const float* as1w     = (const float*)d_in[3];
  const float* ad1w     = (const float*)d_in[4];
  const float* b1       = (const float*)d_in[5];
  const float* W2       = (const float*)d_in[6];
  const float* as2w     = (const float*)d_in[7];
  const float* ad2w     = (const float*)d_in[8];
  const float* b2       = (const float*)d_in[9];

  int N = in_sizes[0] / 128;
  int E = in_sizes[1] / 2;
  int C = in_sizes[9];          // 40
  const long long* e0 = ei;
  const long long* e1 = ei + E;

  // workspace layout (floats); zeroed accumulators are contiguous at the end
  float* ws   = (float*)d_ws;
  float* h1   = ws;                                 // N*64 (later reused as h)
  float* as1  = h1  + (size_t)N * 64;               // N*8
  float* ad1  = as1 + (size_t)N * 8;                // N*8
  float* h2   = ad1 + (size_t)N * 8;                // N*C
  float* as2  = h2  + (size_t)N * C;                // N
  float* ad2  = as2 + (size_t)N;                    // N
  float* s1   = ad2 + (size_t)N;                    // N*8   (zero block start)
  float* agg1 = s1  + (size_t)N * 8;                // N*64
  float* s2   = agg1 + (size_t)N * 64;              // N
  float* agg2 = s2  + (size_t)N;                    // N*C

  const int T = 256;
  auto blocks = [](long n, int t) { return dim3((unsigned)((n + t - 1) / t)); };

  // 1) zero accumulators (s1, agg1, s2, agg2)
  long zn = (long)N * (8 + 64 + 1 + C);
  k_zero<<<blocks(zn, T), T, 0, stream>>>(s1, zn);

  // 2) layer-1 GEMM (WMMA)
  k_gemm1<<<dim3((N + 127) / 128), T, 0, stream>>>(x, W1, h1, N);

  // 3) attention logits per node
  k_att1<<<blocks((long)N * 8, T), T, 0, stream>>>(h1, as1w, ad1w, as1, ad1, N);

  // 4) softmax denominators (self-loops guarantee non-empty segments)
  long e8 = (long)(E + N) * 8;
  k_esum1<<<blocks(e8, T), T, 0, stream>>>(e0, e1, E, N, as1, ad1, s1);

  // 5) weighted aggregation
  k_eagg1<<<blocks(e8, T), T, 0, stream>>>(e0, e1, E, N, as1, ad1, s1, h1, agg1);

  // 6) elu(agg1 + b1) -> reuse h1 buffer as h
  k_elu<<<blocks((long)N * 64, T), T, 0, stream>>>(agg1, b1, h1, (long)N * 64);

  // 7) layer-2 GEMM (WMMA, padded N-dim)
  k_gemm2<<<dim3((N + 127) / 128), T, 0, stream>>>(h1, W2, h2, N, C);

  // 8) layer-2 attention logits
  k_att2<<<blocks((long)N, T), T, 0, stream>>>(h2, as2w, ad2w, as2, ad2, N, C);

  // 9) layer-2 softmax denominators
  k_esum2<<<blocks((long)(E + N), T), T, 0, stream>>>(e0, e1, E, N, as2, ad2, s2);

  // 10) layer-2 aggregation (8-channel chunks)
  int chunks = (C + 7) / 8;
  k_eagg2<<<blocks((long)(E + N) * chunks, T), T, 0, stream>>>(
      e0, e1, E, N, as2, ad2, s2, h2, agg2, C, chunks);

  // 11) log_softmax -> d_out
  k_lsm<<<blocks((long)N, T), T, 0, stream>>>(agg2, b2, (float*)d_out, N, C);
}
